// MoE_58188216926723
// MI455X (gfx1250) — matrix operand order repaired and checked by reading, not compiled, not run on hardware
//
#include <hip/hip_runtime.h>

#define B_TOK 4096
#define DDIM  1024
#define HDIM  4096
#define NEXP  16
#define TOPK  4
#define LDA   40   // padded LDS row pitch in bf16 elems (80B: conflict-free, 16B-aligned)

typedef __bf16 v16bf __attribute__((ext_vector_type(16)));
typedef __bf16 v8bf  __attribute__((ext_vector_type(8)));
typedef __bf16 v4bf  __attribute__((ext_vector_type(4)));
typedef float  v8f   __attribute__((ext_vector_type(8)));
typedef float  v4f   __attribute__((ext_vector_type(4)));

// Build a 16-elem bf16 fragment from two contiguous 8-elem runs (2x ds_load_b128).
__device__ __forceinline__ v16bf frag2(const __bf16* p0, const __bf16* p1) {
    v8bf lo = *(const v8bf*)p0;
    v8bf hi = *(const v8bf*)p1;
    return __builtin_shufflevector(lo, hi, 0, 1, 2, 3, 4, 5, 6, 7,
                                   8, 9, 10, 11, 12, 13, 14, 15);
}

// Async copy 16B per lane: global (bf16 data) -> LDS, tracked by ASYNCcnt.
__device__ __forceinline__ void async_copy_b128(__bf16* lds_dst, const __bf16* gsrc) {
    unsigned lo = (unsigned)(size_t)lds_dst;   // low 32 bits of generic ptr == LDS offset
    asm volatile("global_load_async_to_lds_b128 %0, %1, off"
                 :: "v"(lo), "v"(gsrc) : "memory");
}

__device__ __forceinline__ void wait_async() {
    asm volatile("s_wait_asynccnt 0" ::: "memory");
}

__global__ void MoE_zero(float* __restrict__ out, int n, int* __restrict__ counts) {
    int i = blockIdx.x * blockDim.x + threadIdx.x;
    if (i < NEXP) counts[i] = 0;
    for (; i < n; i += gridDim.x * blockDim.x) out[i] = 0.0f;
}

// x (f32) -> xbf (bf16), one pass, vectorized.
__global__ void MoE_cvt_x(const float* __restrict__ x, __bf16* __restrict__ xbf, int n4) {
    int stride = gridDim.x * blockDim.x;
    for (int t = blockIdx.x * blockDim.x + threadIdx.x; t < n4; t += stride) {
        v4f v = *(const v4f*)&x[(size_t)t * 4];
        *(v4bf*)&xbf[(size_t)t * 4] = __builtin_convertvector(v, v4bf);
    }
}

// One wave32 per token: lanes 0-15 -> clean logits, lanes 16-31 -> noise logits.
__global__ __launch_bounds__(256) void MoE_gate(
    const float* __restrict__ x, const float* __restrict__ noise,
    const float* __restrict__ w_gate, const float* __restrict__ w_noise,
    const float* __restrict__ b_noise,
    int* __restrict__ counts, int* __restrict__ tokList, float* __restrict__ gateList)
{
    const int lane = threadIdx.x & 31;
    const int wv   = threadIdx.x >> 5;
    const int b    = blockIdx.x * 8 + wv;
    const int e    = lane & 15;

    const float* xr = x + (size_t)b * DDIM;
    const float* w  = (lane < 16) ? w_gate : w_noise;
    float acc = 0.0f;
    for (int d = 0; d < DDIM; ++d)
        acc = fmaf(xr[d], w[d * NEXP + e], acc);

    const float other = __shfl_xor(acc, 16, 32);  // lanes<16 receive noise logit
    float noisy;
    if (lane < 16) {
        float sp_in = other + b_noise[e];
        float sp    = sp_in > 20.0f ? sp_in : log1pf(__expf(sp_in));  // softplus
        noisy = acc + noise[(size_t)b * NEXP + e] * sp;
    } else {
        noisy = -__builtin_inff();
    }

    // rank among 16 experts (tie-break: lower index wins, matches lax.top_k)
    int rank = 0;
#pragma unroll
    for (int j = 0; j < 16; ++j) {
        float vj = __shfl(noisy, j, 32);
        if ((vj > noisy) || (vj == noisy && j < e)) rank++;
    }
    const bool sel = (lane < 16) && (rank < TOPK);

    float m = sel ? noisy : -__builtin_inff();
#pragma unroll
    for (int off = 8; off > 0; off >>= 1) m = fmaxf(m, __shfl_xor(m, off, 32));
    float ex = sel ? __expf(noisy - m) : 0.0f;
    float s = ex;
#pragma unroll
    for (int off = 8; off > 0; off >>= 1) s += __shfl_xor(s, off, 32);

    if (sel) {
        float gate = ex / s;
        int pos = atomicAdd(&counts[e], 1);
        tokList[e * B_TOK + pos]  = b;
        gateList[e * B_TOK + pos] = gate;
    }
}

__global__ void MoE_prefix(const int* __restrict__ counts, int* __restrict__ offsets) {
    if (threadIdx.x == 0) {
        int s = 0;
        for (int i = 0; i < NEXP; ++i) { offsets[i] = s; s += counts[i]; }
    }
}

// h = relu(x[tokens] @ W1[e] + b1[e]) in bf16, compact slot layout.
// Block tile 128x128, 8 waves (2 row-tiles x 4 col-tiles each), LDS double-buffered,
// A staged via global_load_async_to_lds_b128, B transpose-converted f32->bf16.
__global__ __launch_bounds__(256) void MoE_gemm1(
    const __bf16* __restrict__ xbf, const float* __restrict__ W1,
    const float* __restrict__ b1,
    const int* __restrict__ counts, const int* __restrict__ offsets,
    const int* __restrict__ tokList, __bf16* __restrict__ h)
{
    const int e    = blockIdx.z;
    const int ht   = blockIdx.y;
    const int row0 = blockIdx.x * 128;
    const int cnt  = counts[e];
    if (row0 >= cnt) return;
    const int off_e = offsets[e];

    const int tid  = threadIdx.x;
    const int lane = tid & 31;
    const int wave = tid >> 5;
    const int rt0  = (wave & 3) * 2;       // first of 2 row-tiles
    const int cb   = (wave >> 2) * 64;     // column base (4 col-tiles)

    __shared__ __align__(16) __bf16 sA[2][128 * LDA];   // [m][k]
    __shared__ __align__(16) __bf16 sB[2][128 * LDA];   // transposed: [n][k]
    __shared__ int stok[128];

    if (tid < 128) {
        int r = row0 + tid;
        if (r >= cnt) r = cnt - 1;          // padded rows recompute a valid row
        stok[tid] = tokList[e * B_TOK + r];
    }
    __syncthreads();

    v8f acc[2][4];
#pragma unroll
    for (int rp = 0; rp < 2; ++rp)
#pragma unroll
        for (int t = 0; t < 4; ++t)
#pragma unroll
            for (int j = 0; j < 8; ++j) acc[rp][t][j] = 0.0f;

    const float* W1e = W1 + (size_t)e * DDIM * HDIM + (size_t)ht * 128;
    const int k0 = (lane >> 4) * 8;      // A K-half select
    const int kb = (lane >> 4) * 16;     // unused; B also uses k0

    auto stageA = [&](int kc, __bf16* dst) {
#pragma unroll
        for (int i = 0; i < 2; ++i) {
            int task = tid + i * 256;
            int m = task & 127, ko = task >> 7;          // ko: 0..3, 8 elems
            async_copy_b128(&dst[m * LDA + ko * 8],
                            &xbf[(size_t)stok[m] * DDIM + kc + ko * 8]);
        }
    };
    auto stageB = [&](int kc, __bf16* dst) {
#pragma unroll
        for (int i = 0; i < 4; ++i) {
            int task = tid + i * 256;
            int n = task & 127, kq = task >> 7;
            v4f wvv;
#pragma unroll
            for (int j = 0; j < 4; ++j)
                wvv[j] = W1e[(size_t)(kc + kq * 4 + j) * HDIM + n];
            *(v4bf*)&dst[n * LDA + kq * 4] = __builtin_convertvector(wvv, v4bf);
        }
    };

    const int NC = DDIM / 32;
    stageA(0, sA[0]);
    stageB(0, sB[0]);

    for (int c = 0; c < NC; ++c) {
        wait_async();          // our async writes into buffer c complete
        __syncthreads();       // everyone's staging of buffer c visible
        if (c + 2 < NC)        // prefetch weights two chunks ahead
            __builtin_prefetch(&W1e[(size_t)((c + 2) * 32 + (tid >> 5)) * HDIM + (tid & 31) * 4], 0, 1);
        if (c + 1 < NC) {
            stageA((c + 1) * 32, sA[(c + 1) & 1]);
            stageB((c + 1) * 32, sB[(c + 1) & 1]);
        }
        const __bf16* bufA = sA[c & 1];
        const __bf16* bufB = sB[c & 1];

        v16bf afr[2];
#pragma unroll
        for (int rp = 0; rp < 2; ++rp) {
            const __bf16* arow = &bufA[((rt0 + rp) * 16 + (lane & 15)) * LDA];
            afr[rp] = frag2(arow + k0, arow + 16 + k0);
        }
#pragma unroll
        for (int t = 0; t < 4; ++t) {
            const __bf16* brow = &bufB[(cb + t * 16 + (lane & 15)) * LDA];
            v16bf bfr = frag2(brow + k0, brow + 16 + k0);
#pragma unroll
            for (int rp = 0; rp < 2; ++rp)
                acc[rp][t] = __builtin_amdgcn_wmma_f32_16x16x32_bf16(
                    false, afr[rp], false, bfr, (short)0, acc[rp][t], false, false);
        }
    }

    const int half = lane >> 4;
#pragma unroll
    for (int rp = 0; rp < 2; ++rp) {
#pragma unroll
        for (int t = 0; t < 4; ++t) {
            const int hcol   = ht * 128 + cb + t * 16 + (lane & 15);
            const float bias = b1[e * HDIM + hcol];
#pragma unroll
            for (int v = 0; v < 8; ++v) {
                int m_local = (rt0 + rp) * 16 + v + 8 * half;
                int r = row0 + m_local;
                if (r < cnt) {
                    float val = acc[rp][t][v] + bias;
                    val = val > 0.0f ? val : 0.0f;
                    h[(size_t)(off_e + r) * HDIM + hcol] = (__bf16)val;
                }
            }
        }
    }
}

// out[token] += gate * (h @ W2[e] + b2[e]); grid: (B/128, D/128, E).
__global__ __launch_bounds__(256) void MoE_gemm2(
    const __bf16* __restrict__ h, const float* __restrict__ W2,
    const float* __restrict__ b2,
    const int* __restrict__ counts, const int* __restrict__ offsets,
    const int* __restrict__ tokList, const float* __restrict__ gateList,
    float* __restrict__ out)
{
    const int e    = blockIdx.z;
    const int dt   = blockIdx.y;
    const int row0 = blockIdx.x * 128;
    const int cnt  = counts[e];
    if (row0 >= cnt) return;
    const int off_e = offsets[e];

    const int tid  = threadIdx.x;
    const int lane = tid & 31;
    const int wave = tid >> 5;
    const int rt0  = (wave & 3) * 2;
    const int cb   = (wave >> 2) * 64;

    __shared__ __align__(16) __bf16 sA[2][128 * LDA];   // [m][k]
    __shared__ __align__(16) __bf16 sB[2][128 * LDA];   // transposed: [n][k]
    __shared__ int   sslot[128];
    __shared__ int   stok[128];
    __shared__ float sgate[128];

    if (tid < 128) {
        int r  = row0 + tid;
        int rc = r < cnt ? r : cnt - 1;
        sslot[tid] = off_e + rc;
        stok[tid]  = tokList[e * B_TOK + rc];
        sgate[tid] = gateList[e * B_TOK + rc];
    }
    __syncthreads();

    v8f acc[2][4];
#pragma unroll
    for (int rp = 0; rp < 2; ++rp)
#pragma unroll
        for (int t = 0; t < 4; ++t)
#pragma unroll
            for (int j = 0; j < 8; ++j) acc[rp][t][j] = 0.0f;

    const float* W2e = W2 + (size_t)e * HDIM * DDIM + (size_t)dt * 128;
    const int k0 = (lane >> 4) * 8;
    const int kb = (lane >> 4) * 16;

    auto stageA = [&](int kc, __bf16* dst) {
#pragma unroll
        for (int i = 0; i < 2; ++i) {
            int task = tid + i * 256;
            int m = task & 127, ko = task >> 7;
            async_copy_b128(&dst[m * LDA + ko * 8],
                            &h[(size_t)sslot[m] * HDIM + kc + ko * 8]);
        }
    };
    auto stageB = [&](int kc, __bf16* dst) {
#pragma unroll
        for (int i = 0; i < 4; ++i) {
            int task = tid + i * 256;
            int n = task & 127, kq = task >> 7;
            v4f wvv;
#pragma unroll
            for (int j = 0; j < 4; ++j)
                wvv[j] = W2e[(size_t)(kc + kq * 4 + j) * DDIM + n];
            *(v4bf*)&dst[n * LDA + kq * 4] = __builtin_convertvector(wvv, v4bf);
        }
    };

    const int NC = HDIM / 32;
    stageA(0, sA[0]);
    stageB(0, sB[0]);

    for (int c = 0; c < NC; ++c) {
        wait_async();
        __syncthreads();
        if (c + 2 < NC)
            __builtin_prefetch(&W2e[(size_t)((c + 2) * 32 + (tid >> 5)) * DDIM + (tid & 31) * 4], 0, 1);
        if (c + 1 < NC) {
            stageA((c + 1) * 32, sA[(c + 1) & 1]);
            stageB((c + 1) * 32, sB[(c + 1) & 1]);
        }
        const __bf16* bufA = sA[c & 1];
        const __bf16* bufB = sB[c & 1];

        v16bf afr[2];
#pragma unroll
        for (int rp = 0; rp < 2; ++rp) {
            const __bf16* arow = &bufA[((rt0 + rp) * 16 + (lane & 15)) * LDA];
            afr[rp] = frag2(arow + k0, arow + 16 + k0);
        }
#pragma unroll
        for (int t = 0; t < 4; ++t) {
            const __bf16* brow = &bufB[(cb + t * 16 + (lane & 15)) * LDA];
            v16bf bfr = frag2(brow + k0, brow + 16 + k0);
#pragma unroll
            for (int rp = 0; rp < 2; ++rp)
                acc[rp][t] = __builtin_amdgcn_wmma_f32_16x16x32_bf16(
                    false, afr[rp], false, bfr, (short)0, acc[rp][t], false, false);
        }
    }

    const int half = lane >> 4;
#pragma unroll
    for (int rp = 0; rp < 2; ++rp) {
#pragma unroll
        for (int t = 0; t < 4; ++t) {
            const int dcol   = dt * 128 + cb + t * 16 + (lane & 15);
            const float bias = b2[e * DDIM + dcol];
#pragma unroll
            for (int v = 0; v < 8; ++v) {
                int m_local = (rt0 + rp) * 16 + v + 8 * half;
                int r = row0 + m_local;
                if (r < cnt) {
                    float val = sgate[m_local] * (acc[rp][t][v] + bias);
                    atomicAdd(&out[(size_t)stok[m_local] * DDIM + dcol], val);
                }
            }
        }
    }
}

extern "C" void kernel_launch(void* const* d_in, const int* in_sizes, int n_in,
                              void* d_out, int out_size, void* d_ws, size_t ws_size,
                              hipStream_t stream) {
    (void)in_sizes; (void)n_in; (void)out_size; (void)ws_size;
    const float* x       = (const float*)d_in[0];
    const float* noise   = (const float*)d_in[1];
    const float* w_gate  = (const float*)d_in[2];
    const float* w_noise = (const float*)d_in[3];
    const float* b_noise = (const float*)d_in[4];
    const float* W1      = (const float*)d_in[5];
    const float* b1      = (const float*)d_in[6];
    const float* W2      = (const float*)d_in[7];
    const float* b2      = (const float*)d_in[8];
    float* out = (float*)d_out;

    char* ws = (char*)d_ws;
    int*   counts   = (int*)ws;                                         // 16 ints
    int*   offsets  = (int*)(ws + 256);                                 // 16 ints
    int*   tokList  = (int*)(ws + 512);                                 // 16*4096 ints
    float* gateList = (float*)(ws + 512 + (size_t)NEXP * B_TOK * 4);    // 16*4096 f32
    __bf16* xbf     = (__bf16*)(ws + 512 + (size_t)NEXP * B_TOK * 8);   // 4096*1024 bf16
    __bf16* h       = (__bf16*)(ws + 512 + (size_t)NEXP * B_TOK * 8
                                + (size_t)B_TOK * DDIM * 2);            // 16384*4096 bf16

    MoE_zero<<<1024, 256, 0, stream>>>(out, B_TOK * DDIM, counts);
    MoE_cvt_x<<<1024, 256, 0, stream>>>(x, xbf, B_TOK * DDIM / 4);
    MoE_gate<<<B_TOK / 8, 256, 0, stream>>>(x, noise, w_gate, w_noise, b_noise,
                                            counts, tokList, gateList);
    MoE_prefix<<<1, 32, 0, stream>>>(counts, offsets);
    MoE_gemm1<<<dim3(B_TOK / 128, HDIM / 128, NEXP), 256, 0, stream>>>(
        xbf, W1, b1, counts, offsets, tokList, h);
    MoE_gemm2<<<dim3(B_TOK / 128, DDIM / 128, NEXP), 256, 0, stream>>>(
        h, W2, b2, counts, offsets, tokList, gateList, out);
}
